// SimpleGAT_14139032338581
// MI455X (gfx1250) — matrix operand order repaired and checked by reading, not compiled, not run on hardware
//
#include <hip/hip_runtime.h>
#include <hip/hip_bf16.h>
#include <math.h>

#define NN   50000
#define EE   400000
#define EPx  (EE + NN)     // edges incl. self loops
#define NPAD 50048         // 391 blocks * 128 rows

typedef __attribute__((ext_vector_type(16))) _Float16 v16h;
typedef __attribute__((ext_vector_type(8)))  _Float16 v8h;
typedef __attribute__((ext_vector_type(8)))  float    v8f;

// ---------------- helpers ----------------

__device__ __forceinline__ void atomicMaxF(float* a, float v) {
  // sign-split trick: monotone int ordering for floats
  if (v >= 0.f) atomicMax((int*)a, __float_as_int(v));
  else          atomicMin((unsigned int*)a, __float_as_uint(v));
}

__global__ void k_fill(float* __restrict__ p, float v, long n) {
  long i = (long)blockIdx.x * blockDim.x + threadIdx.x;
  if (i < n) p[i] = v;
}

// ---------------- self-loop weights (fill_value='mean') ----------------

__global__ void k_deg(const int* __restrict__ ei, const float* __restrict__ ew,
                      float* __restrict__ cnt, float* __restrict__ sumw) {
  int k = blockIdx.x * blockDim.x + threadIdx.x;
  if (k >= EE) return;
  int d = ei[EE + k];
  atomicAdd(cnt + d, 1.f);
  atomicAdd(sumw + d, ew[k]);
}

__global__ void k_loopw(const float* __restrict__ cnt, float* __restrict__ sumw) {
  int i = blockIdx.x * blockDim.x + threadIdx.x;
  if (i >= NN) return;
  sumw[i] = sumw[i] / fmaxf(cnt[i], 1.f);   // in-place -> loop weight
}

// ---------------- f32 -> f16 activation conversion (layer-1 input) ----------------

__global__ void k_cvt128(const float* __restrict__ x, _Float16* __restrict__ xh) {
  long i = (long)blockIdx.x * blockDim.x + threadIdx.x;
  if (i >= (long)NPAD * 128) return;
  long row = i >> 7;
  _Float16 v = (_Float16)0.f;
  if (row < NN) v = (_Float16)x[i];
  xh[i] = v;
}

// ---------------- weight repack: f32 [din,dout] -> WMMA B fragments (f16) ----------------
// Wp[((kt*NT+nt)*32 + lane)*16 + j] = W[kt*32 + (j>>3)*16 + (lane>>4)*8 + (j&7)][nt*16 + (lane&15)]

__global__ void k_pack(const float* __restrict__ W, _Float16* __restrict__ Wp,
                       int din, int dout, int KT, int NT) {
  int total = KT * NT * 32 * 16;
  int idx = blockIdx.x * blockDim.x + threadIdx.x;
  if (idx >= total) return;
  int j    = idx & 15;
  int lane = (idx >> 4) & 31;
  int g    = idx >> 9;              // kt*NT + nt
  int nt   = g % NT;
  int kt   = g / NT;
  int n = lane & 15, hi = lane >> 4;
  int k   = kt * 32 + (j >> 3) * 16 + hi * 8 + (j & 7);
  int col = nt * 16 + n;
  float v = 0.f;
  if (k < din && col < dout) v = W[(long)k * dout + col];
  Wp[idx] = (_Float16)v;
}

// ---------------- WMMA GEMM: Y[nrows,NT*16] = Xh[nrows,ldx] * W + bias ----------------

template <int KT, int NT>
__global__ __launch_bounds__(256)
void k_gemm(const _Float16* __restrict__ Xh, int ldx,
            const _Float16* __restrict__ Wp,
            const float* __restrict__ bias,
            float* __restrict__ Y, int ldy, int nrows) {
  int wave = threadIdx.x >> 5, lane = threadIdx.x & 31;
  int mbase = (blockIdx.x * 8 + wave) * 16;
  if (mbase >= nrows) return;
  int ml = lane & 15, hi = lane >> 4;

  v8f z = {0.f, 0.f, 0.f, 0.f, 0.f, 0.f, 0.f, 0.f};
  v8f acc[NT];
#pragma unroll
  for (int t = 0; t < NT; ++t) acc[t] = z;

  const _Float16* xrow = Xh + (long)(mbase + ml) * ldx + hi * 8;
#pragma unroll
  for (int kt = 0; kt < KT; ++kt) {
    v8h a0 = *(const v8h*)(xrow + kt * 32);        // K = kt*32 + hi*8 + 0..7
    v8h a1 = *(const v8h*)(xrow + kt * 32 + 16);   // K = kt*32 + 16 + hi*8 + 0..7
    v16h a;
#pragma unroll
    for (int i = 0; i < 8; ++i) { a[i] = a0[i]; a[i + 8] = a1[i]; }
#pragma unroll
    for (int nt = 0; nt < NT; ++nt) {
      v16h b = *(const v16h*)(Wp + ((long)(kt * NT + nt) * 32 + lane) * 16);
      acc[nt] = __builtin_amdgcn_wmma_f32_16x16x32_f16(
          false, a, false, b, (short)0, acc[nt], false, false);
    }
  }
#pragma unroll
  for (int nt = 0; nt < NT; ++nt) {
    float bs = bias[nt * 16 + ml];
#pragma unroll
    for (int r = 0; r < 8; ++r) {
      int row = mbase + hi * 8 + r;                // D: lanes 0-15 -> M=r, 16-31 -> M=8+r
      if (row < nrows) Y[(long)row * ldy + nt * 16 + ml] = acc[nt][r] + bs;
    }
  }
}

// ---------------- edge pass 1: logits + segment max ----------------

template <int CH, int HD, int CC>
__global__ void k_edge1(const int* __restrict__ ei, const float* __restrict__ ew,
                        const float* __restrict__ loopw,
                        const float* __restrict__ xl, const float* __restrict__ xr,
                        const float* __restrict__ We, const float* __restrict__ att,
                        float* __restrict__ logits, float* __restrict__ lmax) {
  int k = blockIdx.x * blockDim.x + threadIdx.x;
  if (k >= EPx) return;
  int s, d; float w;
  if (k < EE) { s = ei[k]; d = ei[EE + k]; w = ew[k]; }
  else        { s = k - EE; d = s; w = loopw[s]; }

  const float4* pa = (const float4*)(xl + (long)s * CH);
  const float4* pb = (const float4*)(xr + (long)d * CH);
  const float4* pe = (const float4*)We;
  const float4* pt = (const float4*)att;

  float lg[HD];
#pragma unroll
  for (int h = 0; h < HD; ++h) lg[h] = 0.f;

#pragma unroll
  for (int j = 0; j < CH / 4; ++j) {
    float4 a = pa[j], b = pb[j], e4 = pe[j], t4 = pt[j];
    int h = (4 * j) / CC;   // CC % 4 == 0 -> whole chunk in one head
    float v;
    v = a.x + b.x + w * e4.x; v = v > 0.f ? v : 0.2f * v; lg[h] += t4.x * v;
    v = a.y + b.y + w * e4.y; v = v > 0.f ? v : 0.2f * v; lg[h] += t4.y * v;
    v = a.z + b.z + w * e4.z; v = v > 0.f ? v : 0.2f * v; lg[h] += t4.z * v;
    v = a.w + b.w + w * e4.w; v = v > 0.f ? v : 0.2f * v; lg[h] += t4.w * v;
  }
#pragma unroll
  for (int h = 0; h < HD; ++h) {
    logits[(long)k * HD + h] = lg[h];
    atomicMaxF(lmax + (long)d * HD + h, lg[h]);
  }
}

// ---------------- edge pass 2: exp, denom, numerator scatter ----------------

template <int CH, int HD, int CC>
__global__ void k_edge2(const int* __restrict__ ei,
                        const float* __restrict__ xl,
                        const float* __restrict__ logits, const float* __restrict__ lmax,
                        float* __restrict__ denom, float* __restrict__ agg) {
  int k = blockIdx.x * blockDim.x + threadIdx.x;
  if (k >= EPx) return;
  int s, d;
  if (k < EE) { s = ei[k]; d = ei[EE + k]; }
  else        { s = k - EE; d = s; }

  float ea[HD];
#pragma unroll
  for (int h = 0; h < HD; ++h) {
    ea[h] = __expf(logits[(long)k * HD + h] - lmax[(long)d * HD + h]);
    atomicAdd(denom + (long)d * HD + h, ea[h]);
  }
  const float4* pa = (const float4*)(xl + (long)s * CH);
  float* po = agg + (long)d * CH;
#pragma unroll
  for (int j = 0; j < CH / 4; ++j) {
    float4 a = pa[j];
    float sc = ea[(4 * j) / CC];
    atomicAdd(po + 4 * j + 0, sc * a.x);
    atomicAdd(po + 4 * j + 1, sc * a.y);
    atomicAdd(po + 4 * j + 2, sc * a.z);
    atomicAdd(po + 4 * j + 3, sc * a.w);
  }
}

// ---------------- fused node pass: /denom + bias, LayerNorm, (+res), ReLU ----------------

__global__ void k_fin_mid(const float* __restrict__ agg, const float* __restrict__ denom,
                          const float* __restrict__ bias,
                          const float* __restrict__ g, const float* __restrict__ bb,
                          const float* __restrict__ res,
                          float* __restrict__ xo, _Float16* __restrict__ xh) {
  int i = blockIdx.x * blockDim.x + threadIdx.x;
  if (i >= NPAD) return;
  if (i >= NN) {                       // zero padded rows of the f16 activation buffer
#pragma unroll
    for (int c = 0; c < 96; ++c) xh[(long)i * 96 + c] = (_Float16)0.f;
    return;
  }
  float inv[4];
#pragma unroll
  for (int h = 0; h < 4; ++h) inv[h] = 1.f / denom[(long)i * 4 + h];

  float v[80]; float sum = 0.f;
#pragma unroll
  for (int c = 0; c < 80; ++c) {
    float t = agg[(long)i * 80 + c] * inv[c / 20] + bias[c];
    v[c] = t; sum += t;
  }
  float mu = sum * (1.f / 80.f);
  float s2 = 0.f;
#pragma unroll
  for (int c = 0; c < 80; ++c) { float dd = v[c] - mu; s2 += dd * dd; }
  float r = rsqrtf(s2 * (1.f / 80.f) + 1e-5f);

#pragma unroll
  for (int c = 0; c < 80; ++c) {
    float t = (v[c] - mu) * r * g[c] + bb[c];
    if (res) t += res[(long)i * 80 + c];
    t = fmaxf(t, 0.f);
    xo[(long)i * 80 + c] = t;
    xh[(long)i * 96 + c] = (_Float16)t;
  }
#pragma unroll
  for (int c = 80; c < 96; ++c) xh[(long)i * 96 + c] = (_Float16)0.f;  // K pad
}

__global__ void k_fin4(const float* __restrict__ agg, const float* __restrict__ denom,
                       const float* __restrict__ bias, float* __restrict__ y) {
  int i = blockIdx.x * blockDim.x + threadIdx.x;
  if (i >= NN) return;
  float inv = 1.f / denom[i];
#pragma unroll
  for (int c = 0; c < 16; ++c) y[(long)i * 16 + c] = agg[(long)i * 16 + c] * inv + bias[c];
}

// ---------------- host driver ----------------

static inline void fillf(float* p, float v, long n, hipStream_t s) {
  k_fill<<<dim3((unsigned)((n + 255) / 256)), dim3(256), 0, s>>>(p, v, n);
}

extern "C" void kernel_launch(void* const* d_in, const int* in_sizes, int n_in,
                              void* d_out, int out_size, void* d_ws, size_t ws_size,
                              hipStream_t stream) {
  (void)in_sizes; (void)n_in; (void)out_size; (void)ws_size;
  const float* x  = (const float*)d_in[0];
  const int*   ei = (const int*)d_in[1];
  const float* ew = (const float*)d_in[2];
  auto F = [&](int i) { return (const float*)d_in[i]; };
  // flattened param order: l1..l4 {Wl,bl,Wr,br,We,att,b} at 3..30, ln{1..3}{g,b} at 31..36

  // ---- workspace carve (256B-aligned regions) ----
  float* ws = (float*)d_ws;
  size_t off = 0;
  auto alloc = [&](size_t nf) { float* p = ws + off; off += (nf + 63) & ~(size_t)63; return p; };
  float* cnt    = alloc(NN);
  float* loopw  = alloc(NN);
  float* lmax   = alloc(4L * NN);
  float* denom  = alloc(4L * NN);
  float* logits = alloc(4L * EPx);
  float* xl     = alloc(80L * NN);
  float* xr     = alloc(80L * NN);
  float* agg    = alloc(80L * NN);
  float* xa     = alloc(80L * NN);
  float* xb     = alloc(80L * NN);
  _Float16* xh  = (_Float16*)alloc((long)NPAD * 64);  // NPAD*128 halves
  _Float16* wpA = (_Float16*)alloc(5120);             // 10240 halves
  _Float16* wpB = (_Float16*)alloc(5120);

  const int TB = 256;
  const unsigned gE  = (EE + TB - 1) / TB;
  const unsigned gEP = (EPx + TB - 1) / TB;
  const unsigned gN  = (NN + TB - 1) / TB;
  const unsigned gNP = (NPAD + TB - 1) / TB;
  const unsigned gG  = NPAD / 128;                    // 391 blocks, 8 waves * 16 rows

  // ---- self-loop weights (scatter-mean of edge_weight by dst) ----
  fillf(cnt, 0.f, NN, stream);
  fillf(loopw, 0.f, NN, stream);
  k_deg<<<gE, TB, 0, stream>>>(ei, ew, cnt, loopw);
  k_loopw<<<gN, TB, 0, stream>>>(cnt, loopw);

  // ---- layer-1 input to f16 ----
  k_cvt128<<<(unsigned)(((long)NPAD * 128 + TB - 1) / TB), TB, 0, stream>>>(x, xh);

  // ---- layers 1..3 (80-wide, 4 heads x 20) ----
  for (int L = 0; L < 3; ++L) {
    int base = 3 + L * 7;
    const float *Wl = F(base), *bl = F(base + 1), *Wr = F(base + 2), *br = F(base + 3);
    const float *We = F(base + 4), *att = F(base + 5), *bo = F(base + 6);
    const float *lng = F(31 + L * 2), *lnb = F(32 + L * 2);
    int din = (L == 0) ? 128 : 80;
    int ldx = (L == 0) ? 128 : 96;
    int KT  = (L == 0) ? 4 : 3;
    unsigned gPk = (unsigned)((KT * 5 * 512 + TB - 1) / TB);

    k_pack<<<gPk, TB, 0, stream>>>(Wl, wpA, din, 80, KT, 5);
    k_pack<<<gPk, TB, 0, stream>>>(Wr, wpB, din, 80, KT, 5);
    if (L == 0) {
      k_gemm<4, 5><<<gG, TB, 0, stream>>>(xh, ldx, wpA, bl, xl, 80, NN);
      k_gemm<4, 5><<<gG, TB, 0, stream>>>(xh, ldx, wpB, br, xr, 80, NN);
    } else {
      k_gemm<3, 5><<<gG, TB, 0, stream>>>(xh, ldx, wpA, bl, xl, 80, NN);
      k_gemm<3, 5><<<gG, TB, 0, stream>>>(xh, ldx, wpB, br, xr, 80, NN);
    }
    fillf(lmax, -INFINITY, 4L * NN, stream);
    fillf(denom, 0.f, 4L * NN, stream);
    fillf(agg, 0.f, 80L * NN, stream);
    k_edge1<80, 4, 20><<<gEP, TB, 0, stream>>>(ei, ew, loopw, xl, xr, We, att, logits, lmax);
    k_edge2<80, 4, 20><<<gEP, TB, 0, stream>>>(ei, xl, logits, lmax, denom, agg);

    const float* res = (L == 0) ? nullptr : ((L == 1) ? xa : xb);
    float* xout      = (L == 0) ? xa : ((L == 1) ? xb : xa);
    k_fin_mid<<<gNP, TB, 0, stream>>>(agg, denom, bo, lng, lnb, res, xout, xh);
  }

  // ---- layer 4 (16-wide, 1 head) ----
  {
    int base = 24;
    const float *Wl = F(base), *bl = F(base + 1), *Wr = F(base + 2), *br = F(base + 3);
    const float *We = F(base + 4), *att = F(base + 5), *bo = F(base + 6);
    unsigned gPk = (unsigned)((3 * 1 * 512 + TB - 1) / TB);
    k_pack<<<gPk, TB, 0, stream>>>(Wl, wpA, 80, 16, 3, 1);
    k_pack<<<gPk, TB, 0, stream>>>(Wr, wpB, 80, 16, 3, 1);
    k_gemm<3, 1><<<gG, TB, 0, stream>>>(xh, 96, wpA, bl, xl, 16, NN);
    k_gemm<3, 1><<<gG, TB, 0, stream>>>(xh, 96, wpB, br, xr, 16, NN);
    fillf(lmax, -INFINITY, NN, stream);
    fillf(denom, 0.f, NN, stream);
    fillf(agg, 0.f, 16L * NN, stream);
    k_edge1<16, 1, 16><<<gEP, TB, 0, stream>>>(ei, ew, loopw, xl, xr, We, att, logits, lmax);
    k_edge2<16, 1, 16><<<gEP, TB, 0, stream>>>(ei, xl, logits, lmax, denom, agg);
    k_fin4<<<gN, TB, 0, stream>>>(agg, denom, bo, (float*)d_out);
  }
}
